// HypQKVAttention_18021682774999
// MI455X (gfx1250) — matrix operand order repaired and checked by reading, not compiled, not run on hardware
//
#include <hip/hip_runtime.h>

typedef __attribute__((ext_vector_type(16))) _Float16 v16h;
typedef __attribute__((ext_vector_type(8)))  _Float16 v8h;
typedef __attribute__((ext_vector_type(4)))  _Float16 v4h;
typedef __attribute__((ext_vector_type(8)))  float    v8f;
typedef __attribute__((ext_vector_type(4)))  int      v4i;

#define WMMA_F16(a, b, c) \
  __builtin_amdgcn_wmma_f32_16x16x32_f16(false, (a), false, (b), (short)0, (c), false, false)

#if defined(__gfx1250__) && __has_builtin(__builtin_amdgcn_global_load_async_to_lds_b128) && \
    __has_builtin(__builtin_amdgcn_s_wait_asynccnt)
#define USE_ASYNC_LDS 1
// exact param types per hipcc diagnostic: (v4i __device__*, v4i __local*, imm, imm)
typedef __attribute__((address_space(1))) v4i* g_v4i_p;
typedef __attribute__((address_space(3))) v4i* l_v4i_p;
#else
#define USE_ASYNC_LDS 0
#endif

// ---------------------------------------------------------------------------
// A-matrix (16x32, f16): lane L -> row (L&15); halves 0-15 = K[16*hi..15+16*hi]
// (hi = L>>4). Row-major source, leading dim ld.
__device__ __forceinline__ v16h load_A16x32(const _Float16* p, int ld, int lane) {
  const int m = lane & 15, hi = lane >> 4;
  const _Float16* r = p + m * ld + hi * 16;
  v8h lo = *(const v8h*)r;
  v8h hh = *(const v8h*)(r + 8);
  v16h a;
#pragma unroll
  for (int i = 0; i < 8; ++i) { a[i] = lo[i]; a[i + 8] = hh[i]; }
  return a;
}

// B-matrix (32x16, f16): lane L -> column (L&15), K = 16*(L>>4)..+15.
// Source layout: B[k][n] = p[n*ld + k]  (column contiguous over k).
__device__ __forceinline__ v16h load_B32x16(const _Float16* p, int ld, int lane) {
  const int n = lane & 15, hi = lane >> 4;
  return *(const v16h*)(p + n * ld + hi * 16);
}

// ---------------------------------------------------------------------------
__global__ void cvt_f32_f16(const float* __restrict__ a, _Float16* __restrict__ o, int n) {
  int i = blockIdx.x * blockDim.x + threadIdx.x;
  int stride = gridDim.x * blockDim.x;
  for (; i < n; i += stride) o[i] = (_Float16)a[i];
}

// ---------------------------------------------------------------------------
// Per-row geometry transform. One block of 256 threads per row of 768 floats.
__global__ void transform_branch(const float* __restrict__ x, _Float16* __restrict__ xb,
                                 int branch) {
  __shared__ float red[8];
  const int row = blockIdx.x;
  const float* xr = x + (size_t)row * 768;
  float v0 = xr[threadIdx.x];
  float v1 = xr[threadIdx.x + 256];
  float v2 = xr[threadIdx.x + 512];
  float ss = v0 * v0 + v1 * v1 + v2 * v2;
#pragma unroll
  for (int off = 16; off >= 1; off >>= 1) ss += __shfl_xor(ss, off, 32);
  if ((threadIdx.x & 31) == 0) red[threadIdx.x >> 5] = ss;
  __syncthreads();
  float tot = red[0] + red[1] + red[2] + red[3] + red[4] + red[5] + red[6] + red[7];
  float norm = fmaxf(sqrtf(tot), 1e-5f);
  const float sc = 0.31622776601683794f;  // sqrt(0.1)
  float f;
  if (branch == 0) {
    f = 1.0f;
  } else if (branch == 1) {
    float t = tanhf(sc * norm);
    t = fminf(t, 1.0f - 4e-3f);            // expmap0 factor fused with project clamp
    f = t / (sc * norm);
  } else {
    float arg = fminf(fmaxf(sc * norm, -1.0f + 1e-5f), 1.0f - 1e-5f);
    float at = 0.5f * logf((1.0f + arg) / (1.0f - arg));
    f = at / (sc * norm);
  }
  _Float16* o = xb + (size_t)row * 768;
  o[threadIdx.x]       = (_Float16)(v0 * f);
  o[threadIdx.x + 256] = (_Float16)(v1 * f);
  o[threadIdx.x + 512] = (_Float16)(v2 * f);
}

// ---------------------------------------------------------------------------
// out[M,N] = A[M,K] * W[N,K]^T + bias. One wave -> 32x64 tile, 2x4 WMMA
// accumulators, K stepped by 32.
__global__ void gemm_bias_f16(const _Float16* __restrict__ A, const _Float16* __restrict__ W,
                              const float* __restrict__ bias, _Float16* __restrict__ out,
                              int M, int N, int K) {
  const int wave = (int)((blockIdx.x * blockDim.x + threadIdx.x) >> 5);
  const int lane = threadIdx.x & 31;
  const int ntiles = N / 64;
  const int m0 = (wave / ntiles) * 32;
  const int n0 = (wave % ntiles) * 64;
  if (m0 >= M) return;

  v8f c[2][4];
#pragma unroll
  for (int i = 0; i < 2; ++i)
#pragma unroll
    for (int t = 0; t < 4; ++t) c[i][t] = (v8f){};

  const _Float16* a0p = A + (size_t)m0 * K;
  const _Float16* a1p = A + (size_t)(m0 + 16) * K;
  for (int k0 = 0; k0 < K; k0 += 32) {
    v16h a0 = load_A16x32(a0p + k0, K, lane);
    v16h a1 = load_A16x32(a1p + k0, K, lane);
#pragma unroll
    for (int t = 0; t < 4; ++t) {
      v16h b = load_B32x16(W + (size_t)(n0 + t * 16) * K + k0, K, lane);
      c[0][t] = WMMA_F16(a0, b, c[0][t]);
      c[1][t] = WMMA_F16(a1, b, c[1][t]);
    }
  }
  const int col = lane & 15, rhi = (lane >> 4) * 8;
#pragma unroll
  for (int t = 0; t < 4; ++t) {
    const float bc = bias[n0 + t * 16 + col];
#pragma unroll
    for (int i = 0; i < 2; ++i)
#pragma unroll
      for (int j = 0; j < 8; ++j)
        out[(size_t)(m0 + i * 16 + rhi + j) * N + n0 + t * 16 + col] =
            (_Float16)(c[i][t][j] + bc);
  }
}

// ---------------------------------------------------------------------------
// Flash attention per (batch, head, 64-query tile). 4 waves/block, each wave
// owns 16 query rows. Online softmax; no 1/sqrt(dh) scaling (faithful to ref).
// V tile is double-buffered into LDS with CDNA5 async loads when available.
#define VSTR 80   // padded row stride (halves) of Vt: 160B, 32B aligned, conflict-breaking
#define PSTR 72   // padded row stride (halves) of P tiles
#define RAWN (64 * 64)

__global__ void flash_attn(const _Float16* __restrict__ q, const _Float16* __restrict__ k,
                           const _Float16* __restrict__ v, float* __restrict__ out,
                           int accumulate) {
  const int Ntok = 1024, D = 768, H = 12, dh = 64;
  const int tile = blockIdx.x;
  const int qt = tile & 15;
  const int bh = tile >> 4;
  const int h = bh % H, b = bh / H;
  const int waveId = threadIdx.x >> 5, lane = threadIdx.x & 31;
  const int m0 = qt * 64 + waveId * 16;

  __shared__ __attribute__((aligned(64))) _Float16 Vt[64 * VSTR];      // V^T tile [d][key]
  __shared__ __attribute__((aligned(64))) _Float16 Plds[4][16 * PSTR]; // per-wave P tile
#if USE_ASYNC_LDS
  __shared__ __attribute__((aligned(64))) _Float16 raw[2][RAWN];       // raw V tiles [key][d]
#endif

  const _Float16* qbase = q + (size_t)(b * Ntok) * D + h * dh;
  const _Float16* kbase = k + (size_t)(b * Ntok) * D + h * dh;
  const _Float16* vbase = v + (size_t)(b * Ntok) * D + h * dh;

  const v16h A0 = load_A16x32(qbase + (size_t)m0 * D, D, lane);
  const v16h A1 = load_A16x32(qbase + (size_t)m0 * D + 32, D, lane);

#if USE_ASYNC_LDS
  // prologue: kick tile kt=0 into raw[0] (4 x b128 per thread = 8KB tile)
#pragma unroll
  for (int i = 0; i < 4; ++i) {
    const int cch = threadIdx.x * 4 + i;
    const int key = cch >> 3, dp = (cch & 7) * 8;
    __builtin_amdgcn_global_load_async_to_lds_b128(
        (g_v4i_p)(vbase + (size_t)key * D + dp),
        (l_v4i_p)(&raw[0][key * 64 + dp]), 0, 0);
  }
#endif

  v8f O[4] = {};
  float mrow[8], lrow[8];
#pragma unroll
  for (int j = 0; j < 8; ++j) { mrow[j] = -1e30f; lrow[j] = 0.0f; }

  for (int kt = 0; kt < Ntok; kt += 64) {
    const int buf = (kt >> 6) & 1;
    __syncthreads();  // (a) everyone done with Vt / raw from previous iteration

#if USE_ASYNC_LDS
    {   // prefetch next tile into the other buffer, then wait for current one
      const int ktn = (kt + 64) & (Ntok - 1);
#pragma unroll
      for (int i = 0; i < 4; ++i) {
        const int cch = threadIdx.x * 4 + i;
        const int key = cch >> 3, dp = (cch & 7) * 8;
        __builtin_amdgcn_global_load_async_to_lds_b128(
            (g_v4i_p)(vbase + (size_t)(ktn + key) * D + dp),
            (l_v4i_p)(&raw[buf ^ 1][key * 64 + dp]), 0, 0);
      }
      __builtin_amdgcn_s_wait_asynccnt(4);  // previous batch (current buf) complete
    }
    __syncthreads();  // (b) raw[buf] visible to all waves
#endif

    {   // transpose into Vt: 8 keys x 4 d per thread; b64 reads, b128 LDS stores
      const int keyblk = (threadIdx.x & 7) * 8;
      const int dblk = (threadIdx.x >> 3) * 4;
      v4h r[8];
#pragma unroll
      for (int kk = 0; kk < 8; ++kk) {
#if USE_ASYNC_LDS
        r[kk] = *(const v4h*)(&raw[buf][(keyblk + kk) * 64 + dblk]);
#else
        r[kk] = *(const v4h*)(vbase + (size_t)(kt + keyblk + kk) * D + dblk);
#endif
      }
#pragma unroll
      for (int d = 0; d < 4; ++d) {
        v8h pk;
#pragma unroll
        for (int kk = 0; kk < 8; ++kk) pk[kk] = r[kk][d];
        *(v8h*)(Vt + (dblk + d) * VSTR + keyblk) = pk;
      }
    }
    __syncthreads();  // (c) Vt ready

    // S = Q K^T : 16 rows x 64 keys (4 N-tiles, K=dh=64 in two steps)
    v8f S[4];
#pragma unroll
    for (int t = 0; t < 4; ++t) {
      const _Float16* kb = kbase + (size_t)(kt + t * 16) * D;
      v16h b0 = load_B32x16(kb, D, lane);
      v16h b1 = load_B32x16(kb + 32, D, lane);
      v8f c = {};
      c = WMMA_F16(A0, b0, c);
      c = WMMA_F16(A1, b1, c);
      S[t] = c;
    }

    // online softmax (row reductions stay inside each 16-lane half)
#pragma unroll
    for (int j = 0; j < 8; ++j) {
      float mx = fmaxf(fmaxf(S[0][j], S[1][j]), fmaxf(S[2][j], S[3][j]));
#pragma unroll
      for (int off = 8; off >= 1; off >>= 1) mx = fmaxf(mx, __shfl_xor(mx, off, 32));
      const float mnew = fmaxf(mrow[j], mx);
      const float corr = __expf(mrow[j] - mnew);
      mrow[j] = mnew;
      lrow[j] *= corr;
#pragma unroll
      for (int t = 0; t < 4; ++t) O[t][j] *= corr;
      float rs = 0.0f;
#pragma unroll
      for (int t = 0; t < 4; ++t) {
        const float p = __expf(S[t][j] - mnew);
        S[t][j] = p;
        rs += p;
      }
#pragma unroll
      for (int off = 8; off >= 1; off >>= 1) rs += __shfl_xor(rs, off, 32);
      lrow[j] += rs;
    }

    // C-layout -> A-layout shuffle of P via per-wave LDS tile
    _Float16* Pw = Plds[waveId];
    const int col = lane & 15, rhi = (lane >> 4) * 8;
#pragma unroll
    for (int t = 0; t < 4; ++t)
#pragma unroll
      for (int j = 0; j < 8; ++j)
        Pw[(rhi + j) * PSTR + t * 16 + col] = (_Float16)S[t][j];
    __syncthreads();  // (d) order P stores before A-layout reads

    const v16h PA0 = load_A16x32(Pw, PSTR, lane);
    const v16h PA1 = load_A16x32(Pw + 32, PSTR, lane);
#pragma unroll
    for (int t = 0; t < 4; ++t) {
      v16h b0 = load_B32x16(Vt + (t * 16) * VSTR, VSTR, lane);
      v16h b1 = load_B32x16(Vt + (t * 16) * VSTR + 32, VSTR, lane);
      O[t] = WMMA_F16(PA0, b0, O[t]);
      O[t] = WMMA_F16(PA1, b1, O[t]);
    }
  }

  // epilogue: normalize and accumulate into f32 output
  float* obase = out + ((size_t)(b * Ntok) + m0) * D + h * dh;
  const int col = lane & 15, rhi = (lane >> 4) * 8;
#pragma unroll
  for (int t = 0; t < 4; ++t)
#pragma unroll
    for (int j = 0; j < 8; ++j) {
      const float val = O[t][j] / lrow[j];
      float* p = obase + (size_t)(rhi + j) * D + t * 16 + col;
      if (accumulate) *p += val; else *p = val;
    }
}

// ---------------------------------------------------------------------------
extern "C" void kernel_launch(void* const* d_in, const int* in_sizes, int n_in,
                              void* d_out, int out_size, void* d_ws, size_t ws_size,
                              hipStream_t stream) {
  const float* x   = (const float*)d_in[0];
  const float* w_q = (const float*)d_in[1];
  const float* b_q = (const float*)d_in[2];
  const float* w_k = (const float*)d_in[3];
  const float* b_k = (const float*)d_in[4];
  const float* w_v = (const float*)d_in[5];
  const float* b_v = (const float*)d_in[6];
  float* out = (float*)d_out;

  const int B = 8, Ntok = 1024, D = 768;
  const int M = B * Ntok;                 // 8192
  const size_t MD = (size_t)M * D;        // 6,291,456 elements
  const size_t WN = (size_t)3 * D * D;    // 1,769,472 per weight tensor

  char* ws = (char*)d_ws;
  _Float16* wq16 = (_Float16*)ws; ws += WN * 2;
  _Float16* wk16 = (_Float16*)ws; ws += WN * 2;
  _Float16* wv16 = (_Float16*)ws; ws += WN * 2;
  _Float16* xb16 = (_Float16*)ws; ws += MD * 2;
  _Float16* q16  = (_Float16*)ws; ws += MD * 2;
  _Float16* k16  = (_Float16*)ws; ws += MD * 2;
  _Float16* v16  = (_Float16*)ws; ws += MD * 2;

  cvt_f32_f16<<<1024, 256, 0, stream>>>(w_q, wq16, (int)WN);
  cvt_f32_f16<<<1024, 256, 0, stream>>>(w_k, wk16, (int)WN);
  cvt_f32_f16<<<1024, 256, 0, stream>>>(w_v, wv16, (int)WN);

  const int gemm_blocks = (M / 32) * (D / 64) / 8;  // 8 waves per 256-thread block
  for (int br = 0; br < 3; ++br) {
    transform_branch<<<M, 256, 0, stream>>>(x, xb16, br);
    gemm_bias_f16<<<gemm_blocks, 256, 0, stream>>>(xb16, wq16 + (size_t)br * D * D,
                                                   b_q + br * D, q16, M, D, D);
    gemm_bias_f16<<<gemm_blocks, 256, 0, stream>>>(xb16, wk16 + (size_t)br * D * D,
                                                   b_k + br * D, k16, M, D, D);
    gemm_bias_f16<<<gemm_blocks, 256, 0, stream>>>(xb16, wv16 + (size_t)br * D * D,
                                                   b_v + br * D, v16, M, D, D);
    flash_attn<<<B * 12 * 16, 128, 0, stream>>>(q16, k16, v16, out, br > 0);
  }
}
